// CausalSelfAttention_74594991997031
// MI455X (gfx1250) — compile-verified, matrix operand order checked by reading, not run on hardware
//
#include <hip/hip_runtime.h>
#include <hip/hip_bf16.h>
#include <stdint.h>

// ---------------------------------------------------------------------------
// Causal self-attention for MI455X (gfx1250, wave32, WMMA bf16 16x16x32).
// Pipeline: f32->bf16 casts -> QKV GEMM (software-pipelined, scatter epilogue,
// V stored transposed) -> RoPE -> flash attention (TDM double-buffered K/V
// staging to LDS, online softmax, WMMA S and PV) -> out-proj GEMM (fp32 out).
// ---------------------------------------------------------------------------

typedef __attribute__((ext_vector_type(16))) __bf16 v16bf;
typedef __attribute__((ext_vector_type(8)))  float  v8f;
typedef __attribute__((ext_vector_type(4)))  unsigned int v4u;
typedef __attribute__((ext_vector_type(8)))  int    v8i;
typedef __attribute__((ext_vector_type(4)))  int    v4i;

#define WMMA_BF16(a, b, c)                                                     \
  __builtin_amdgcn_wmma_f32_16x16x32_bf16(false, (a), false, (b), (short)0,    \
                                          (c), false, false)

#if __has_builtin(__builtin_amdgcn_tensor_load_to_lds) &&                      \
    __has_builtin(__builtin_amdgcn_s_wait_tensorcnt)
#define HAVE_TDM 1
#else
#define HAVE_TDM 0
#endif

static constexpr int BB = 4;
static constexpr int TT = 2048;
static constexpr int CC = 1024;
static constexpr int HH = 16;
static constexpr int DD = 64;
static constexpr int MROWS = BB * TT;  // 8192
static constexpr int N3 = 3 * CC;      // 3072

union FragU {
  v16bf bf;
  uint32_t u[8];
};

// Load one 16x32 bf16 WMMA fragment from a row-major source:
// lane l -> free-dim row (l & 15); contraction dwords follow the ISA layout
// (lanes 0-15: K=0..7,16..23 ; lanes 16-31: K=8..15,24..31), i.e. dword
// offsets {kh+0..3, kh+8..11} with kh = (l>>4)*4.
__device__ inline v16bf load_frag(const uint32_t* __restrict__ src,
                                  int stride_u, int lane) {
  const int row = lane & 15;
  const int kh = (lane >> 4) << 2;
  FragU f;
  const uint32_t* p = src + (size_t)row * stride_u + kh;
#pragma unroll
  for (int j = 0; j < 4; ++j) {
    f.u[j] = p[j];
    f.u[j + 4] = p[j + 8];
  }
  return f.bf;
}

#if HAVE_TDM
// Issue one TDM tensor_load_to_lds for a (tile1 x tile0) bf16 tile (tile1==0
// -> 1D of tile0 elements).  All args must be wave-uniform.  D# layout per
// CDNA5 ISA 8.3/8.4: group0 = {count, lds_addr, global_addr lo/hi | type=2},
// group1 = {data_size=2B, tensor_dim0/1, tile_dim0/1, tensor_dim0_stride}.
// This toolchain exposes the 6-arg builtin: (v4u, v8i, v4i, v4i, v8i, cpol).
__device__ inline void tdm_load_bf16(uint32_t lds_addr, const __bf16* gptr,
                                     uint32_t tile0, uint32_t tile1,
                                     uint32_t tdim0, uint32_t tdim1,
                                     uint32_t stride0) {
  const uint64_t ga = (uint64_t)(uintptr_t)gptr;
  v4u g0;
  g0.x = 1u;                                     // count = 1 valid descriptor
  g0.y = lds_addr;                               // LDS byte address
  g0.z = (uint32_t)ga;                           // global_addr[31:0]
  g0.w = (uint32_t)(ga >> 32) | (2u << 30);      // global_addr[56:32], type=2
  v8i g1;
  g1[0] = (int)(1u << 16);                       // data_size = 1 -> 2 bytes
  g1[1] = (int)((tdim0 & 0xFFFFu) << 16);        // tensor_dim0[15:0]
  g1[2] = (int)((tdim0 >> 16) | ((tdim1 & 0xFFFFu) << 16));
  g1[3] = (int)((tdim1 >> 16) | (tile0 << 16));  // tile_dim0
  g1[4] = (int)(tile1 & 0xFFFFu);                // tile_dim1 (0 => 1D)
  g1[5] = (int)stride0;                          // tensor_dim0_stride[31:0]
  g1[6] = 0;
  g1[7] = 0;
  const v4i z4 = {0, 0, 0, 0};                   // groups 2/3 unused (<=2D)
  const v8i z8 = {0, 0, 0, 0, 0, 0, 0, 0};
  __builtin_amdgcn_tensor_load_to_lds(g0, g1, z4, z4, z8, 0);
}
#endif

// ---------------------------------------------------------------------------
__global__ void __launch_bounds__(256) f32_to_bf16_kernel(
    const float* __restrict__ in, __bf16* __restrict__ out, int n) {
  int i = blockIdx.x * blockDim.x + threadIdx.x;
  if (i < n) out[i] = (__bf16)in[i];
}

// ---------------------------------------------------------------------------
// Software-pipelined 16x64 GEMM tile body: prefetch step k+1 fragments before
// the 4 WMMAs of step k so VMEM overlaps the matrix pipe.  unroll 2 turns the
// (a,b) <- (an,bn) rotation into register renaming (no v_mov chains).
__device__ inline void gemm_tile_16x64(const uint32_t* __restrict__ A,
                                       const uint32_t* __restrict__ B,
                                       int stride_u, int ksteps, int lane,
                                       v8f cf[4]) {
  v16bf a = load_frag(A, stride_u, lane);
  v16bf b[4];
#pragma unroll
  for (int t4 = 0; t4 < 4; ++t4)
    b[t4] = load_frag(B + (size_t)(t4 * 16) * stride_u, stride_u, lane);

#pragma unroll 2
  for (int k = 0; k < ksteps - 1; ++k) {
    v16bf an = load_frag(A + (k + 1) * 16, stride_u, lane);
    v16bf bn[4];
#pragma unroll
    for (int t4 = 0; t4 < 4; ++t4)
      bn[t4] = load_frag(B + (size_t)(t4 * 16) * stride_u + (k + 1) * 16,
                         stride_u, lane);
#pragma unroll
    for (int t4 = 0; t4 < 4; ++t4) cf[t4] = WMMA_BF16(a, b[t4], cf[t4]);
    a = an;
#pragma unroll
    for (int t4 = 0; t4 < 4; ++t4) b[t4] = bn[t4];
  }
#pragma unroll
  for (int t4 = 0; t4 < 4; ++t4) cf[t4] = WMMA_BF16(a, b[t4], cf[t4]);
}

// ---------------------------------------------------------------------------
// QKV projection: qkv[m,n] = sum_c xb[m,c] * wab[n,c].  One wave -> one 16x64
// tile; scatter epilogue into q[b,h,t,d], k[b,h,t,d], vt[b,h,d,t].
__global__ void __launch_bounds__(256) qkv_gemm_kernel(
    const __bf16* __restrict__ xb, const __bf16* __restrict__ wab,
    __bf16* __restrict__ qb, __bf16* __restrict__ kb,
    __bf16* __restrict__ vtb) {
  const int lane = threadIdx.x & 31;
  const int wid = blockIdx.x * 8 + (threadIdx.x >> 5);
  const int NT = N3 / 64;  // 48 column tiles
  const int mt = wid / NT, nt = wid % NT;
  const int m0 = mt * 16, n0 = nt * 64;

  const uint32_t* A = (const uint32_t*)xb + (size_t)m0 * (CC / 2);
  const uint32_t* B = (const uint32_t*)wab + (size_t)n0 * (CC / 2);

  v8f cf[4] = {};
  gemm_tile_16x64(A, B, CC / 2, CC / 32, lane, cf);

  // C layout: lane l, vgpr r holds (M = r + 8*(l>>4), N = l&15).
  const int half8 = (lane >> 4) << 3;
  const int cl = lane & 15;
#pragma unroll
  for (int t4 = 0; t4 < 4; ++t4) {
    const int n = n0 + t4 * 16 + cl;
    const int part = n >> 10;  // 0=Q 1=K 2=V
    const int rem = n & 1023;
    const int h = rem >> 6, d = rem & 63;
#pragma unroll
    for (int r = 0; r < 8; ++r) {
      const int m = m0 + half8 + r;
      const int b = m >> 11, tt = m & 2047;
      const __bf16 val = (__bf16)cf[t4][r];
      const size_t bh = (size_t)(b * HH + h);
      if (part == 0)
        qb[(bh * TT + tt) * DD + d] = val;
      else if (part == 1)
        kb[(bh * TT + tt) * DD + d] = val;
      else
        vtb[(bh * DD + d) * TT + tt] = val;
    }
  }
}

// ---------------------------------------------------------------------------
// RoPE in place on a [B,H,T,D] bf16 tensor; one thread per (row, pair).
__global__ void __launch_bounds__(256) rope_kernel(
    __bf16* __restrict__ t, const float* __restrict__ fc,
    const float* __restrict__ fs, int npairs) {
  int i = blockIdx.x * blockDim.x + threadIdx.x;
  if (i >= npairs) return;
  const int p = i & 31;     // pair index within head dim
  const int rest = i >> 5;  // (b*H + h)*T + tt
  const int tt = rest & 2047;
  const float c = fc[tt * 32 + p];
  const float s = fs[tt * 32 + p];
  const size_t off = (size_t)rest * DD + 2 * p;
  const float xr = (float)t[off];
  const float xi = (float)t[off + 1];
  t[off] = (__bf16)(xr * c - xi * s);
  t[off + 1] = (__bf16)(xr * s + xi * c);
}

// ---------------------------------------------------------------------------
// Flash attention: 8 waves/block, each wave owns 16 query rows of one (b,h).
// K (32x64, contiguous => 1D) and V^T (64 rows x 32 cols, stride T => 2D)
// blocks are staged into per-wave LDS slices by the Tensor Data Mover,
// double-buffered: issue TDM for block j+1, s_wait_tensorcnt 2 => block j
// ready.  Online softmax in fp32; P re-fragmented through LDS.
__global__ void __launch_bounds__(256) attn_kernel(
    const __bf16* __restrict__ qb, const __bf16* __restrict__ kb,
    const __bf16* __restrict__ vtb, __bf16* __restrict__ yb) {
  __shared__ __align__(16) __bf16 lds_k[8][2][32 * DD];
  __shared__ __align__(16) __bf16 lds_v[8][2][DD * 32];
  __shared__ __align__(16) __bf16 lds_p[8][16 * 32];

  const int lane = threadIdx.x & 31;
  const int wave = threadIdx.x >> 5;
  const int bh = blockIdx.y;                    // b*H + h
  const int t0 = blockIdx.x * 128 + wave * 16;  // first query row of wave
  const int half8 = (lane >> 4) << 3;           // row offset 0 / 8
  const int col = lane & 15;                    // N index of this lane

  // Q tile 16x64 kept in registers (two K-half fragments).
  const uint32_t* qrow = (const uint32_t*)(qb + ((size_t)bh * TT + t0) * DD);
  const v16bf qf0 = load_frag(qrow, DD / 2, lane);
  const v16bf qf1 = load_frag(qrow + 16, DD / 2, lane);

  v8f of[4] = {};
  float mrow[8], srow[8];
#pragma unroll
  for (int r = 0; r < 8; ++r) {
    mrow[r] = -1e30f;
    srow[r] = 0.0f;
  }
  const float scale = 0.125f;  // 1/sqrt(64)
  const int kv_lim = t0 + 16;  // exclusive bound on kv0

  const __bf16* kbase = kb + (size_t)bh * TT * DD;
  const __bf16* vbase = vtb + (size_t)bh * DD * TT;

#if HAVE_TDM
  const uint32_t ldsk0 = (uint32_t)(uintptr_t)&lds_k[wave][0][0];
  const uint32_t ldsv0 = (uint32_t)(uintptr_t)&lds_v[wave][0][0];
  // Prologue: stage block 0 into buffer 0.
  tdm_load_bf16(ldsk0, kbase, 32 * DD, 0, 1u << 22, 1, 32 * DD);
  tdm_load_bf16(ldsv0, vbase, 32, DD, TT, DD, TT);
#endif

  for (int kv0 = 0; kv0 < kv_lim; kv0 += 32) {
    const int buf = (kv0 >> 5) & 1;
    const __bf16* kblk;
    const __bf16* vblk;
#if HAVE_TDM
    if (kv0 + 32 < kv_lim) {  // issue next block into other buffer, then wait
      const int nb = buf ^ 1;
      tdm_load_bf16(ldsk0 + (uint32_t)(nb * 32 * DD * 2),
                    kbase + (size_t)(kv0 + 32) * DD, 32 * DD, 0, 1u << 22, 1,
                    32 * DD);
      tdm_load_bf16(ldsv0 + (uint32_t)(nb * DD * 32 * 2), vbase + kv0 + 32,
                    32, DD, TT, DD, TT);
      __builtin_amdgcn_s_wait_tensorcnt((short)2);
    } else {
      __builtin_amdgcn_s_wait_tensorcnt((short)0);
    }
    asm volatile("" ::: "memory");
    kblk = &lds_k[wave][buf][0];
    vblk = &lds_v[wave][buf][0];
#else
    kblk = kbase + (size_t)kv0 * DD;
    vblk = vbase + kv0;  // note: row stride TT in this fallback
#endif
    const int ks = DD / 2;                  // K row stride (dwords)
    const int vs = HAVE_TDM ? 16 : TT / 2;  // V row stride (dwords)

    // S = Q * K^T for a 16x32 block (two 16x16 tiles, contraction = 64).
    v8f sc0 = {}, sc1 = {};
    {
      const uint32_t* kr0 = (const uint32_t*)kblk;
      sc0 = WMMA_BF16(qf0, load_frag(kr0, ks, lane), sc0);
      sc0 = WMMA_BF16(qf1, load_frag(kr0 + 16, ks, lane), sc0);
      const uint32_t* kr1 = (const uint32_t*)(kblk + (size_t)16 * DD);
      sc1 = WMMA_BF16(qf0, load_frag(kr1, ks, lane), sc1);
      sc1 = WMMA_BF16(qf1, load_frag(kr1 + 16, ks, lane), sc1);
    }

    const bool need_mask = (kv0 + 31 > t0);
#pragma unroll
    for (int r = 0; r < 8; ++r) {
      float v0 = sc0[r] * scale;
      float v1 = sc1[r] * scale;
      if (need_mask) {
        const int row = t0 + half8 + r;
        if (kv0 + col > row) v0 = -1e30f;
        if (kv0 + 16 + col > row) v1 = -1e30f;
      }
      // Row max across the 16 lanes of this half (xor stays within half).
      float bm = fmaxf(v0, v1);
      bm = fmaxf(bm, __shfl_xor(bm, 1));
      bm = fmaxf(bm, __shfl_xor(bm, 2));
      bm = fmaxf(bm, __shfl_xor(bm, 4));
      bm = fmaxf(bm, __shfl_xor(bm, 8));
      const float mn = fmaxf(mrow[r], bm);
      const float alpha = __expf(mrow[r] - mn);
      const float p0 = __expf(v0 - mn);
      const float p1 = __expf(v1 - mn);
      float rs = p0 + p1;
      rs += __shfl_xor(rs, 1);
      rs += __shfl_xor(rs, 2);
      rs += __shfl_xor(rs, 4);
      rs += __shfl_xor(rs, 8);
      srow[r] = srow[r] * alpha + rs;
      mrow[r] = mn;
#pragma unroll
      for (int t4 = 0; t4 < 4; ++t4) of[t4][r] = of[t4][r] * alpha;
      // Stash P row-major [16][32] for re-fragmentation as WMMA A operand.
      const int lrow = half8 + r;
      lds_p[wave][lrow * 32 + col] = (__bf16)p0;
      lds_p[wave][lrow * 32 + 16 + col] = (__bf16)p1;
    }
    // Cross-lane LDS RAW within the wave: drain DS counter before reload.
    asm volatile("s_wait_dscnt 0x0" ::: "memory");
    const v16bf pa = load_frag((const uint32_t*)&lds_p[wave][0], 16, lane);
#pragma unroll
    for (int t4 = 0; t4 < 4; ++t4) {
      const uint32_t* vrow = (const uint32_t*)(vblk + (size_t)(t4 * 16) *
                                                          (HAVE_TDM ? 32 : TT));
      of[t4] = WMMA_BF16(pa, load_frag(vrow, vs, lane), of[t4]);
    }
  }

  // Normalize and write y[b, t, h*64 + d] as bf16.
  const int b = bh >> 4, h = bh & 15;
#pragma unroll
  for (int r = 0; r < 8; ++r) {
    const int trow = t0 + half8 + r;
    const float inv = 1.0f / srow[r];
    const size_t rowoff = ((size_t)(b * TT + trow)) * CC + h * DD;
#pragma unroll
    for (int t4 = 0; t4 < 4; ++t4)
      yb[rowoff + t4 * 16 + col] = (__bf16)(of[t4][r] * inv);
  }
}

// ---------------------------------------------------------------------------
// Output projection: out[m,n] = sum_c yb[m,c] * wpb[n,c], fp32 result.
__global__ void __launch_bounds__(256) proj_gemm_kernel(
    const __bf16* __restrict__ yb, const __bf16* __restrict__ wpb,
    float* __restrict__ out) {
  const int lane = threadIdx.x & 31;
  const int wid = blockIdx.x * 8 + (threadIdx.x >> 5);
  const int NT = CC / 64;  // 16
  const int mt = wid / NT, nt = wid % NT;
  const int m0 = mt * 16, n0 = nt * 64;

  const uint32_t* A = (const uint32_t*)yb + (size_t)m0 * (CC / 2);
  const uint32_t* B = (const uint32_t*)wpb + (size_t)n0 * (CC / 2);

  v8f cf[4] = {};
  gemm_tile_16x64(A, B, CC / 2, CC / 32, lane, cf);

  const int half8 = (lane >> 4) << 3;
  const int cl = lane & 15;
#pragma unroll
  for (int r = 0; r < 8; ++r) {
    const int m = m0 + half8 + r;
#pragma unroll
    for (int t4 = 0; t4 < 4; ++t4)
      out[(size_t)m * CC + n0 + t4 * 16 + cl] = cf[t4][r];
  }
}

// ---------------------------------------------------------------------------
extern "C" void kernel_launch(void* const* d_in, const int* in_sizes, int n_in,
                              void* d_out, int out_size, void* d_ws,
                              size_t ws_size, hipStream_t stream) {
  const float* x = (const float*)d_in[0];
  const float* fc = (const float*)d_in[1];
  const float* fs = (const float*)d_in[2];
  const float* wa = (const float*)d_in[3];
  const float* wp = (const float*)d_in[4];
  float* out = (float*)d_out;

  // Workspace carve-up (~92 MB total, 256B aligned chunks).
  char* ws = (char*)d_ws;
  size_t off = 0;
  auto carve = [&](size_t bytes) -> void* {
    void* p = ws + off;
    off += (bytes + 255) & ~(size_t)255;
    return p;
  };
  __bf16* xb = (__bf16*)carve((size_t)MROWS * CC * 2);
  __bf16* wab = (__bf16*)carve((size_t)N3 * CC * 2);
  __bf16* wpb = (__bf16*)carve((size_t)CC * CC * 2);
  __bf16* qb = (__bf16*)carve((size_t)MROWS * CC * 2);
  __bf16* kb = (__bf16*)carve((size_t)MROWS * CC * 2);
  __bf16* vtb = (__bf16*)carve((size_t)MROWS * CC * 2);
  __bf16* yb = (__bf16*)carve((size_t)MROWS * CC * 2);

  f32_to_bf16_kernel<<<(MROWS * CC + 255) / 256, 256, 0, stream>>>(x, xb,
                                                                   MROWS * CC);
  f32_to_bf16_kernel<<<(N3 * CC + 255) / 256, 256, 0, stream>>>(wa, wab,
                                                                N3 * CC);
  f32_to_bf16_kernel<<<(CC * CC + 255) / 256, 256, 0, stream>>>(wp, wpb,
                                                                CC * CC);

  // 512 row-tiles * 48 col-tiles = 24576 wave tiles, 8 waves per block.
  qkv_gemm_kernel<<<(MROWS / 16) * (N3 / 64) / 8, 256, 0, stream>>>(
      xb, wab, qb, kb, vtb);

  const int npairs = BB * HH * TT * (DD / 2);
  rope_kernel<<<(npairs + 255) / 256, 256, 0, stream>>>(qb, fc, fs, npairs);
  rope_kernel<<<(npairs + 255) / 256, 256, 0, stream>>>(kb, fc, fs, npairs);

  attn_kernel<<<dim3(TT / 128, BB * HH), 256, 0, stream>>>(qb, kb, vtb, yb);

  proj_gemm_kernel<<<(MROWS / 16) * (CC / 64) / 8, 256, 0, stream>>>(yb, wpb,
                                                                     out);
}
